// CoordinateDecoder_5068061409565
// MI455X (gfx1250) — matrix operand order repaired and checked by reading, not compiled, not run on hardware
//
#include <hip/hip_runtime.h>
#include <hip/hip_bf16.h>
#include <math.h>

typedef __attribute__((ext_vector_type(16))) _Float16 v16h;
typedef __attribute__((ext_vector_type(8)))  _Float16 h8;
typedef __attribute__((ext_vector_type(4)))  _Float16 h4;
typedef __attribute__((ext_vector_type(8)))  float    v8f;
typedef __attribute__((ext_vector_type(4)))  float    f4;

#define NFREQ  10
#define BATCH  8
#define GH     64
#define GW     64
#define GC     128
#define NPTS   65536
#define ROWS   128          // samples per workgroup
#define WS     192          // LDS activation/weight row stride (f16 elements)
#define CHUNKS (NPTS / ROWS) // 512

__device__ __forceinline__ float gelu_tanh(float x) {
  float u = 0.7978845608028654f * (x + 0.044715f * x * x * x);
  return 0.5f * x * (1.0f + tanhf(u));
}

// One MLP layer: D[128x128] = gelu(act[128xK] * Wt^T + bias), all operands in LDS.
// act: row-major [128][WS] f16. wt: transposed weights [n][k] row-major [128][WS] f16.
// Each wave computes a 16-row stripe across all 128 output columns (8 WMMA tiles).
template<int K>
__device__ __forceinline__ void mlp_layer(const _Float16* __restrict__ act,
                                          const _Float16* __restrict__ wt,
                                          const float* __restrict__ bias,
                                          _Float16* __restrict__ dst,
                                          int tid) {
  const int lane = tid & 31;
  const int wave = tid >> 5;        // 0..7
  const int m0   = wave * 16;
  const int l15  = lane & 15;
  const int hi   = lane >> 4;       // half-wave select
  const int kbA  = hi * 8;          // A frag: lanes 16-31 hold K+8..15 / K+24..31
  const int kbB  = hi * 8;          // B frag: lanes 16-31 hold K+8..15 / K+24..31

  v8f zero = {0.f, 0.f, 0.f, 0.f, 0.f, 0.f, 0.f, 0.f};
  v8f acc[8];
#pragma unroll
  for (int t = 0; t < 8; ++t) acc[t] = zero;

  const _Float16* arow = act + (m0 + l15) * WS;
#pragma unroll
  for (int k0 = 0; k0 < K; k0 += 32) {
    h8 alo = *(const h8*)(arow + k0 + kbA);        // ds_load_b128
    h8 ahi = *(const h8*)(arow + k0 + kbA + 16);   // ds_load_b128
    v16h A = __builtin_shufflevector(alo, ahi, 0,1,2,3,4,5,6,7,8,9,10,11,12,13,14,15);
#pragma unroll
    for (int nt = 0; nt < 8; ++nt) {
      const _Float16* wp = wt + (nt * 16 + l15) * WS + k0 + kbB;
      h8 blo = *(const h8*)(wp);                   // ds_load_b128
      h8 bhi = *(const h8*)(wp + 16);              // ds_load_b128
      v16h Bf = __builtin_shufflevector(blo, bhi, 0,1,2,3,4,5,6,7,8,9,10,11,12,13,14,15);
      acc[nt] = __builtin_amdgcn_wmma_f32_16x16x32_f16(
          false, A, false, Bf, (short)0, acc[nt], false, false);
    }
  }

  // Epilogue: D layout = lane(0-15): M=r, N=lane; lane(16-31): M=8+r, N=lane-16
#pragma unroll
  for (int nt = 0; nt < 8; ++nt) {
    float bv = bias[nt * 16 + l15];
#pragma unroll
    for (int r = 0; r < 8; ++r) {
      float x = acc[nt][r] + bv;
      dst[(m0 + hi * 8 + r) * WS + nt * 16 + l15] = (_Float16)gelu_tanh(x);
    }
  }
}

__global__ __launch_bounds__(256)
void coorddec_kernel(const float* __restrict__ fg, const float* __restrict__ coords,
                     const float* __restrict__ W0, const float* __restrict__ b0,
                     const float* __restrict__ W1, const float* __restrict__ b1,
                     const float* __restrict__ W2, const float* __restrict__ b2,
                     const float* __restrict__ Wout, const float* __restrict__ bout,
                     float* __restrict__ out) {
  extern __shared__ __align__(16) char smem[];
  _Float16* buf0 = (_Float16*)smem;           // activations / rotating weights
  _Float16* buf1 = buf0 + ROWS * WS;
  _Float16* buf2 = buf1 + ROWS * WS;
  float* sb0 = (float*)(buf2 + ROWS * WS);
  float* sb1 = sb0 + 128;
  float* sb2 = sb1 + 128;
  float* swo = sb2 + 128;                     // 128*3
  float* sbo = swo + 384;                     // 3 (+pad)

  const int tid    = threadIdx.x;
  const int blk    = blockIdx.x;
  const int b      = blk / CHUNKS;
  const int n_base = (blk % CHUNKS) * ROWS;

  // ---- Phase 1: W0^T -> buf2 (with enc/pad column remap), small consts, encode+sample -> buf0
  for (int idx = tid; idx < 128 * WS; idx += 256) {
    int nn = idx / WS, kk = idx - nn * WS;
    float v = 0.f;
    if (kk < 42)                   v = W0[kk * 128 + nn];        // enc rows 0..41
    else if (kk >= 48 && kk < 176) v = W0[(kk - 6) * 128 + nn];  // feature rows 42..169
    buf2[nn * WS + kk] = (_Float16)v;
  }
  for (int i = tid; i < 128; i += 256) { sb0[i] = b0[i]; sb1[i] = b1[i]; sb2[i] = b2[i]; }
  for (int i = tid; i < 384; i += 256) swo[i] = Wout[i];
  if (tid < 3) sbo[tid] = bout[tid];

  {
    const int row  = tid >> 1;
    const int half = tid & 1;
    const int n    = n_base + row;
    const float x0 = coords[2 * n + 0];   // -> H index ("rows")
    const float x1 = coords[2 * n + 1];   // -> W index ("cols")
    _Float16* e = buf0 + row * WS;

    if (half == 0) {
      e[0] = (_Float16)x0; e[1] = (_Float16)x1;
      float f = 3.14159265358979323846f;
      for (int i = 0; i < NFREQ; ++i) {
        float a0 = x0 * f, a1 = x1 * f;
        e[2 + 4 * i + 0] = (_Float16)__sinf(a0);
        e[2 + 4 * i + 1] = (_Float16)__sinf(a1);
        e[2 + 4 * i + 2] = (_Float16)__cosf(a0);
        e[2 + 4 * i + 3] = (_Float16)__cosf(a1);
        f *= 2.0f;
      }
      for (int k = 42; k < 48; ++k) e[k] = (_Float16)0.f;   // pad between enc & features
    } else {
      for (int k = 176; k < 192; ++k) e[k] = (_Float16)0.f; // K padding
    }

    // Bilinear sample (coords in [0, 63) so clamp == reflect here)
    float r = fminf(fmaxf((x0 + 1.f) * 0.5f * (GH - 1), 0.f), (float)(GH - 1));
    float c = fminf(fmaxf((x1 + 1.f) * 0.5f * (GW - 1), 0.f), (float)(GW - 1));
    int ir = (int)r; if (ir > GH - 2) ir = GH - 2;
    int ic = (int)c; if (ic > GW - 2) ic = GW - 2;
    float fr = r - ir, fc = c - ic;
    const float* g00 = fg + (((size_t)b * GH + ir) * GW + ic) * GC + half * 64;
    const float* g01 = g00 + GC;
    const float* g10 = g00 + (size_t)GW * GC;
    const float* g11 = g10 + GC;
    float w00 = (1.f - fr) * (1.f - fc), w01 = (1.f - fr) * fc;
    float w10 = fr * (1.f - fc),         w11 = fr * fc;
    _Float16* s = e + 48 + half * 64;
    for (int ch = 0; ch < 64; ch += 4) {
      f4 va = *(const f4*)(g00 + ch);
      f4 vb = *(const f4*)(g01 + ch);
      f4 vc = *(const f4*)(g10 + ch);
      f4 vd = *(const f4*)(g11 + ch);
      h4 o;
#pragma unroll
      for (int j = 0; j < 4; ++j)
        o[j] = (_Float16)(w00 * va[j] + w01 * vb[j] + w10 * vc[j] + w11 * vd[j]);
      *(h4*)(s + ch) = o;
    }
  }
  __syncthreads();

  // Layer 0: buf0 (K=192) x buf2 -> buf1
  mlp_layer<192>(buf0, buf2, sb0, buf1, tid);
  __syncthreads();

  // W1^T -> buf0 (buf0 is dead now)
  for (int idx = tid; idx < 128 * 128; idx += 256) {
    int nn = idx >> 7, kk = idx & 127;
    buf0[nn * WS + kk] = (_Float16)W1[kk * 128 + nn];
  }
  __syncthreads();
  mlp_layer<128>(buf1, buf0, sb1, buf2, tid);
  __syncthreads();

  // W2^T -> buf1
  for (int idx = tid; idx < 128 * 128; idx += 256) {
    int nn = idx >> 7, kk = idx & 127;
    buf1[nn * WS + kk] = (_Float16)W2[kk * 128 + nn];
  }
  __syncthreads();
  mlp_layer<128>(buf2, buf1, sb2, buf0, tid);
  __syncthreads();

  // Output layer 128->3 + tanh (VALU; 3 cols doesn't warrant WMMA)
  if (tid < ROWS) {
    const _Float16* h3 = buf0 + tid * WS;
    float s0 = sbo[0], s1 = sbo[1], s2 = sbo[2];
    for (int k = 0; k < 128; ++k) {
      float hv = (float)h3[k];
      s0 += hv * swo[k * 3 + 0];
      s1 += hv * swo[k * 3 + 1];
      s2 += hv * swo[k * 3 + 2];
    }
    size_t o = ((size_t)b * NPTS + n_base + tid) * 3;
    out[o + 0] = tanhf(s0);
    out[o + 1] = tanhf(s1);
    out[o + 2] = tanhf(s2);
  }
}

extern "C" void kernel_launch(void* const* d_in, const int* in_sizes, int n_in,
                              void* d_out, int out_size, void* d_ws, size_t ws_size,
                              hipStream_t stream) {
  (void)in_sizes; (void)n_in; (void)out_size; (void)d_ws; (void)ws_size;
  const float* fg     = (const float*)d_in[0];
  const float* coords = (const float*)d_in[1];
  const float* W0     = (const float*)d_in[2];
  const float* b0     = (const float*)d_in[3];
  const float* W1     = (const float*)d_in[4];
  const float* b1     = (const float*)d_in[5];
  const float* W2     = (const float*)d_in[6];
  const float* b2     = (const float*)d_in[7];
  const float* Wout   = (const float*)d_in[8];
  const float* bout   = (const float*)d_in[9];
  float* out = (float*)d_out;

  const size_t smem = (size_t)(3 * ROWS * WS) * 2 /* f16 */
                    + (size_t)(128 * 3 + 384 + 4) * sizeof(float); // ~150.5 KB
  (void)hipFuncSetAttribute((const void*)coorddec_kernel,
                            hipFuncAttributeMaxDynamicSharedMemorySize, (int)smem);
  dim3 grid(BATCH * CHUNKS);   // 4096 workgroups
  coorddec_kernel<<<grid, 256, smem, stream>>>(fg, coords, W0, b0, W1, b1, W2, b2,
                                               Wout, bout, out);
}
